// MSARowAttention_81655918231690
// MI455X (gfx1250) — compile-verified, matrix operand order checked by reading, not run on hardware
//
#include <hip/hip_runtime.h>
#include <hip/hip_bf16.h>
#include <cstdint>

// Shapes (fixed by the reference): N=1, S=128, R=256, M=256, Z=128, h=8, c=32
#define S_DIM 128
#define R_DIM 256
#define M_DIM 256
#define Z_DIM 128
#define HEADS 8
#define CDIM  32
#define SR    (S_DIM * R_DIM)   // 32768
#define SCALE 0.17677669529663687f

typedef __attribute__((ext_vector_type(16))) __bf16 bf16x16;
typedef __attribute__((ext_vector_type(8)))  float  f32x8;

// ---------------------------------------------------------------------------
// CDNA5 async global->LDS copy (16 bytes per lane), tracked by ASYNCcnt.
// lds_addr: LDS byte address (low 32 bits of generic pointer = LDS offset).
// gaddr:    64-bit global byte address.
// ---------------------------------------------------------------------------
__device__ __forceinline__ void async_copy16(uint32_t lds_addr, uint64_t gaddr) {
  asm volatile("global_load_async_to_lds_b128 %0, %1, off"
               :: "v"(lds_addr), "v"(gaddr) : "memory");
}
__device__ __forceinline__ void wait_async0() {
  asm volatile("s_wait_asynccnt 0" ::: "memory");
}

// ---------------------------------------------------------------------------
// WMMA fragment helpers (wave32, 16x16x32 bf16, f32 accum)
// A: 16(M)x32(K), row-major source with stride ld (elements).
// Lane map per CDNA5 ISA 7.12.2: lanes 0-15 -> M=lane, K in {0..7,16..23};
// lanes 16-31 -> M=lane-16, K in {8..15,24..31}.
// ---------------------------------------------------------------------------
__device__ __forceinline__ bf16x16 load_frag_a(const __bf16* base, int ld) {
  const unsigned lane = threadIdx.x & 31u;
  const unsigned m  = lane & 15u;
  const unsigned kb = (lane >> 4) * 8u;
  const __bf16* p = base + m * ld + kb;
  bf16x16 f;
#pragma unroll
  for (int i = 0; i < 8; ++i) { f[i] = p[i]; f[i + 8] = p[16 + i]; }
  return f;
}

// B: 32(K)x16(N), memory row-major [K][N] with stride ld.
__device__ __forceinline__ bf16x16 load_frag_b(const __bf16* base, int ld) {
  const unsigned lane = threadIdx.x & 31u;
  const unsigned n  = lane & 15u;
  const unsigned kb = (lane >> 4) * 8u;
  const __bf16* p = base + kb * ld + n;
  bf16x16 f;
#pragma unroll
  for (int i = 0; i < 8; ++i) { f[i] = p[i * ld]; f[i + 8] = p[(16 + i) * ld]; }
  return f;
}

// ---------------------------------------------------------------------------
// f32 -> bf16 conversion (grid stride)
// ---------------------------------------------------------------------------
__global__ void cvt_f32_bf16(const float* __restrict__ in, __bf16* __restrict__ out, int n) {
  for (int i = blockIdx.x * blockDim.x + threadIdx.x; i < n; i += gridDim.x * blockDim.x)
    out[i] = (__bf16)in[i];
}

// ---------------------------------------------------------------------------
// bias = einsum('rpz,zh->hrp'): pair [R,R,Z] f32 x bproj [Z,h] -> bias [h,R,R]
// 134 MFLOP: plain VALU, one thread per (r,p).
// ---------------------------------------------------------------------------
__global__ void bias_kernel(const float* __restrict__ pair, const float* __restrict__ bproj,
                            float* __restrict__ bias) {
  const int idx = blockIdx.x * blockDim.x + threadIdx.x;   // r*R + p, 65536 total
  const float* pr = pair + (size_t)idx * Z_DIM;
  float acc[HEADS];
#pragma unroll
  for (int hh = 0; hh < HEADS; ++hh) acc[hh] = 0.f;
  for (int z = 0; z < Z_DIM; ++z) {
    const float pv = pr[z];
#pragma unroll
    for (int hh = 0; hh < HEADS; ++hh) acc[hh] = fmaf(pv, bproj[z * HEADS + hh], acc[hh]);
  }
#pragma unroll
  for (int hh = 0; hh < HEADS; ++hh) bias[(size_t)hh * (R_DIM * R_DIM) + idx] = acc[hh];
}

// ---------------------------------------------------------------------------
// Tiled bf16 WMMA GEMM: C[M,N] = A[M,K] @ B[K,N]
// Block = 128 threads (4 waves), 64x64 tile, K stepped by 32.
// Double-buffered LDS fed by async global->LDS copies (ASYNCcnt pipeline):
//   issue tile k+1 -> WMMA on tile k -> s_wait_asynccnt 0 -> barrier.
// MODE 0: f32 out, MODE 1: sigmoid -> f32 out (gate), MODE 2: bf16 out (qkv)
// ---------------------------------------------------------------------------
template <int MODE>
__global__ __launch_bounds__(128)
void gemm_bf16(const __bf16* __restrict__ A, const __bf16* __restrict__ B,
               float* __restrict__ outF, __bf16* __restrict__ outH,
               int M, int N, int K) {
  __shared__ __bf16 lds_a[2][64 * 32];
  __shared__ __bf16 lds_b[2][32 * 64];
  const int tid  = threadIdx.x;
  const int wave = tid >> 5;
  const int row0 = blockIdx.x * 64;
  const int col0 = blockIdx.y * 64;

  // per-thread 16-bf16 (32B) slices of the A (64x32) and B (32x64) tiles
  const int ra = tid >> 1, ca = (tid & 1) * 16;
  const int rb = tid >> 2, cb = (tid & 3) * 16;

  auto issue_tile = [&](int buf, int k0) {
    const uint32_t la = (uint32_t)(uintptr_t)(&lds_a[buf][ra * 32 + ca]);
    const uint64_t ga = (uint64_t)(uintptr_t)(A + (size_t)(row0 + ra) * K + k0 + ca);
    async_copy16(la, ga);
    async_copy16(la + 16, ga + 16);
    const uint32_t lb = (uint32_t)(uintptr_t)(&lds_b[buf][rb * 64 + cb]);
    const uint64_t gb = (uint64_t)(uintptr_t)(B + (size_t)(k0 + rb) * N + col0 + cb);
    async_copy16(lb, gb);
    async_copy16(lb + 16, gb + 16);
  };

  f32x8 acc[4] = {};
  issue_tile(0, 0);
  wait_async0();
  __syncthreads();

  const int nk = K >> 5;
  for (int kt = 0; kt < nk; ++kt) {
    const int cur = kt & 1;
    if (kt + 1 < nk) issue_tile(cur ^ 1, (kt + 1) * 32);   // overlap with compute
    const bf16x16 fa = load_frag_a(&lds_a[cur][wave * 16 * 32], 32);
#pragma unroll
    for (int ct = 0; ct < 4; ++ct) {
      const bf16x16 fb = load_frag_b(&lds_b[cur][ct * 16], 64);
      acc[ct] = __builtin_amdgcn_wmma_f32_16x16x32_bf16(
          false, fa, false, fb, (short)0, acc[ct], false, false);
    }
    wait_async0();       // next tile landed while we did the WMMAs
    __syncthreads();
  }

  const unsigned lane = tid & 31u;
  const unsigned n  = lane & 15u;
  const unsigned mb = (lane >> 4) * 8u;
#pragma unroll
  for (int ct = 0; ct < 4; ++ct) {
#pragma unroll
    for (int r = 0; r < 8; ++r) {
      const size_t idx = (size_t)(row0 + wave * 16 + mb + r) * N + col0 + ct * 16 + n;
      float v = acc[ct][r];
      if (MODE == 1) v = 1.0f / (1.0f + __expf(-v));
      if (MODE == 2) outH[idx] = (__bf16)v;
      else           outF[idx] = v;
    }
  }
}

// ---------------------------------------------------------------------------
// Fused attention per (s,h): sim = q k^T ; softmax(*SCALE folded in) ; + bias ;
// attend = probs @ v ; * gate ; store bf16. sim/probs never touch HBM.
// qkv layout: [(s*R + r) * 768 + h*96 + {q:0..31, k:32..63, v:64..95}] bf16
// Block = 256 threads (8 waves). LDS = 16K (K) + 16K (V) + 32K (sim) = 64 KB.
// ---------------------------------------------------------------------------
__global__ __launch_bounds__(256)
void attn_kernel(const __bf16* __restrict__ qkv, const float* __restrict__ bias,
                 const float* __restrict__ gate, __bf16* __restrict__ att) {
  __shared__ __bf16 k_lds[R_DIM * CDIM];
  __shared__ __bf16 v_lds[R_DIM * CDIM];
  __shared__ float  sim_lds[32 * R_DIM];
  __bf16* probs = (__bf16*)sim_lds;   // reused after softmax (16 KB of the 32 KB)

  const int s = blockIdx.x >> 3;
  const int h = blockIdx.x & 7;
  const int tid  = threadIdx.x;
  const int wave = tid >> 5;

  { // stage K and V rows via async global->LDS (64B each per thread)
    const uint64_t base = (uint64_t)(uintptr_t)(qkv + (size_t)(s * R_DIM + tid) * 768 + h * 96);
    const uint32_t lk = (uint32_t)(uintptr_t)(k_lds + tid * CDIM);
    const uint32_t lv = (uint32_t)(uintptr_t)(v_lds + tid * CDIM);
#pragma unroll
    for (int i = 0; i < 4; ++i) {
      async_copy16(lk + i * 16, base + 64  + i * 16);   // k: +32 bf16 = +64B
      async_copy16(lv + i * 16, base + 128 + i * 16);   // v: +64 bf16 = +128B
    }
  }
  wait_async0();
  __syncthreads();

  for (int rp = 0; rp < 8; ++rp) {
    const int row0 = rp * 32;

    // ---- Phase 1: sim (32 x 256) via WMMA, 8 waves x 4 col-tiles --------
    {
      const int rt  = wave >> 2;        // 0..1 (row tile)
      const int ct0 = (wave & 3) * 4;   // col tile base
      const unsigned lane = tid & 31u;
      const unsigned m  = lane & 15u;
      const unsigned kb = (lane >> 4) * 8u;
      // q A-fragment straight from global (rows row0 + rt*16 + m)
      const __bf16* qp = qkv + (size_t)(s * R_DIM + row0 + rt * 16 + m) * 768 + h * 96 + kb;
      bf16x16 fa;
#pragma unroll
      for (int i = 0; i < 8; ++i) { fa[i] = qp[i]; fa[i + 8] = qp[16 + i]; }
#pragma unroll
      for (int j = 0; j < 4; ++j) {
        const int ct = ct0 + j;
        // B = k^T (KxN = 32x256); memory is k[p][cc] => same pattern as A-frag
        const bf16x16 fb = load_frag_a(k_lds + ct * 16 * CDIM, CDIM);
        f32x8 c = {};
        c = __builtin_amdgcn_wmma_f32_16x16x32_bf16(
            false, fa, false, fb, (short)0, c, false, false);
        const unsigned n  = lane & 15u;
        const unsigned mb = (lane >> 4) * 8u;
#pragma unroll
        for (int r = 0; r < 8; ++r)   // raw store; SCALE folded into softmax
          sim_lds[(rt * 16 + mb + r) * R_DIM + ct * 16 + n] = c[r];
      }
    }
    __syncthreads();

    // ---- Phase 2: softmax over the 256 cols, + bias (post-softmax), bf16 ----
    {
      const int row = tid >> 3;  // 0..31
      const int seg = tid & 7;   // 8 threads/row, 32 cols each (same wave)
      float x[32];
      const float* sr = sim_lds + row * R_DIM + seg * 32;
#pragma unroll
      for (int j = 0; j < 32; ++j) x[j] = sr[j] * SCALE;
      float mx = -1e30f;
#pragma unroll
      for (int j = 0; j < 32; ++j) mx = fmaxf(mx, x[j]);
      mx = fmaxf(mx, __shfl_xor(mx, 1, 8));
      mx = fmaxf(mx, __shfl_xor(mx, 2, 8));
      mx = fmaxf(mx, __shfl_xor(mx, 4, 8));
      float sum = 0.f;
#pragma unroll
      for (int j = 0; j < 32; ++j) { x[j] = __expf(x[j] - mx); sum += x[j]; }
      sum += __shfl_xor(sum, 1, 8);
      sum += __shfl_xor(sum, 2, 8);
      sum += __shfl_xor(sum, 4, 8);
      const float inv = 1.0f / sum;
      const float* bp = bias + ((size_t)h * R_DIM + row0 + row) * R_DIM + seg * 32;
      float pb[32];
#pragma unroll
      for (int j = 0; j < 32; ++j) pb[j] = fmaf(x[j], inv, bp[j]);
      __syncthreads();   // all sim reads done before aliased probs writes
      __bf16* pr = probs + row * R_DIM + seg * 32;
#pragma unroll
      for (int j = 0; j < 32; ++j) pr[j] = (__bf16)pb[j];
    }
    __syncthreads();

    // ---- Phase 3: attend = probs @ V (K=256), * gate, store bf16 --------
    if (wave < 4) {   // wave-uniform branch: EXEC stays all-ones for WMMA
      const int rt = wave >> 1;  // 0..1
      const int ct = wave & 1;   // 0..1 (16 of the 32 c-cols)
      f32x8 acc = {};
#pragma unroll
      for (int kt = 0; kt < 8; ++kt) {
        const bf16x16 fa = load_frag_a(probs + rt * 16 * R_DIM + kt * 32, R_DIM);
        const bf16x16 fb = load_frag_b(v_lds + kt * 32 * CDIM + ct * 16, CDIM);
        acc = __builtin_amdgcn_wmma_f32_16x16x32_bf16(
            false, fa, false, fb, (short)0, acc, false, false);
      }
      const unsigned lane = tid & 31u;
      const unsigned n  = lane & 15u;
      const unsigned mb = (lane >> 4) * 8u;
#pragma unroll
      for (int r = 0; r < 8; ++r) {
        const int mg = row0 + rt * 16 + mb + r;
        const int cg = h * CDIM + ct * 16 + n;
        const size_t idx = (size_t)(s * R_DIM + mg) * M_DIM + cg;
        att[idx] = (__bf16)(acc[r] * gate[idx]);
      }
    }
    __syncthreads();  // probs buffer reused as sim next pass
  }
}

// ---------------------------------------------------------------------------
extern "C" void kernel_launch(void* const* d_in, const int* in_sizes, int n_in,
                              void* d_out, int out_size, void* d_ws, size_t ws_size,
                              hipStream_t stream) {
  (void)in_sizes; (void)n_in; (void)out_size; (void)ws_size;
  const float* msa     = (const float*)d_in[0];  // [1,S,R,M]
  const float* pair    = (const float*)d_in[1];  // [1,R,R,Z]
  const float* gproj   = (const float*)d_in[2];  // [M,M]
  const float* qkvproj = (const float*)d_in[3];  // [M,3M]
  const float* oproj   = (const float*)d_in[4];  // [M,M]
  const float* bproj   = (const float*)d_in[5];  // [Z,h]
  float* out = (float*)d_out;

  char* ws = (char*)d_ws;
  size_t off = 0;
  auto take = [&](size_t bytes) -> char* {
    char* p = ws + off;
    off += (bytes + 255) & ~(size_t)255;
    return p;
  };

  __bf16* msa_bf  = (__bf16*)take((size_t)SR * M_DIM * 2);          // 16.8 MB
  __bf16* gpb     = (__bf16*)take((size_t)M_DIM * M_DIM * 2);       // 128 KB
  __bf16* qpb     = (__bf16*)take((size_t)M_DIM * 3 * M_DIM * 2);   // 384 KB
  __bf16* opb     = (__bf16*)take((size_t)M_DIM * M_DIM * 2);       // 128 KB
  float*  gate_f  = (float*) take((size_t)SR * M_DIM * 4);          // 33.6 MB
  __bf16* qkv_bf  = (__bf16*)take((size_t)SR * 3 * M_DIM * 2);      // 50.3 MB
  float*  bias_f  = (float*) take((size_t)HEADS * R_DIM * R_DIM * 4); // 2.1 MB
  __bf16* att_bf  = (__bf16*)take((size_t)SR * M_DIM * 2);          // 16.8 MB

  // 1) f32 -> bf16 conversions
  cvt_f32_bf16<<<4096, 256, 0, stream>>>(msa, msa_bf, SR * M_DIM);
  cvt_f32_bf16<<<256, 256, 0, stream>>>(gproj, gpb, M_DIM * M_DIM);
  cvt_f32_bf16<<<768, 256, 0, stream>>>(qkvproj, qpb, M_DIM * 3 * M_DIM);
  cvt_f32_bf16<<<256, 256, 0, stream>>>(oproj, opb, M_DIM * M_DIM);

  // 2) pair bias [h,R,R]
  bias_kernel<<<(R_DIM * R_DIM) / 256, 256, 0, stream>>>(pair, bproj, bias_f);

  // 3) gate = sigmoid(msa @ gate_proj)   (f32 out)
  gemm_bf16<1><<<dim3(SR / 64, M_DIM / 64), 128, 0, stream>>>(
      msa_bf, gpb, gate_f, nullptr, SR, M_DIM, M_DIM);

  // 4) qkv = msa @ qkv_proj              (bf16 out)
  gemm_bf16<2><<<dim3(SR / 64, (3 * M_DIM) / 64), 128, 0, stream>>>(
      msa_bf, qpb, nullptr, qkv_bf, SR, 3 * M_DIM, M_DIM);

  // 5) fused attention: one block per (s,h)
  attn_kernel<<<S_DIM * HEADS, 256, 0, stream>>>(qkv_bf, bias_f, gate_f, att_bf);

  // 6) out = attend @ out_proj           (f32 out -> d_out)
  gemm_bf16<0><<<dim3(SR / 64, M_DIM / 64), 128, 0, stream>>>(
      att_bf, opb, out, nullptr, SR, M_DIM, M_DIM);
}
